// SpatialRangeAttentionCUDA_50251117363408
// MI455X (gfx1250) — matrix operand order repaired and checked by reading, not compiled, not run on hardware
//
#include <hip/hip_runtime.h>
#include <math.h>

typedef __attribute__((ext_vector_type(16))) _Float16 v16h;
typedef __attribute__((ext_vector_type(8)))  float    v8f;

#define PIX_TILE 64   // pixels per block in GEMM kernels

__device__ inline unsigned int pack_f16x2(float a, float b) {
  union { _Float16 h[2]; unsigned int u; } t;
  t.h[0] = (_Float16)a; t.h[1] = (_Float16)b;
  return t.u;
}

__device__ inline float sigmoidf_(float x) { return 1.0f / (1.0f + __expf(-x)); }

// ---------------------------------------------------------------------------
// Weight pre-pack: f32 row-major w[Ndim][Kdim] -> f16 in exact WMMA B-fragment
// order (CDNA5 ISA 7.12.2, B 32x16: lane half h, element j holds
// K = j + (j&8) + 8h, matching the A fragment).  Zero-padded to Kpad=32*ksteps,
// Npad=16*ntiles.  Element i = (((ntile*ksteps+ks)*32)+lane)*16 + j.
// ---------------------------------------------------------------------------
__global__ void pack_weights_kernel(const float* __restrict__ w, _Float16* __restrict__ outw,
                                    int Kdim, int Ndim, int ksteps, int ntiles) {
  int total = ntiles * ksteps * 32 * 16;
  for (int i = blockIdx.x * blockDim.x + threadIdx.x; i < total;
       i += gridDim.x * blockDim.x) {
    int j    = i & 15;
    int lane = (i >> 4) & 31;
    int blk  = i >> 9;              // ntile*ksteps + ks
    int ks    = blk % ksteps;
    int ntile = blk / ksteps;
    int N = ntile * 16 + (lane & 15);
    int K = ks * 32 + (j & 7) + ((j & 8) << 1) + ((lane & 16) >> 1);
    float v = (K < Kdim && N < Ndim) ? w[N * Kdim + K] : 0.0f;
    outw[i] = (_Float16)v;
  }
}

// ---------------------------------------------------------------------------
// Wave-level 16x16 GEMM tile, K = 32*KSTEPS.
// A from LDS packed f16 pairs: ldsA word (q,p) holds K=2q (lo) and K=2q+1 (hi)
// for pixel p.  A-fragment (16-bit A 16x32 layout): lane m=lane&15,
// half j -> K = j + (j&8) + 8*(lane>=16); pairs stay within one word:
//   word-row(jj) = ks*16 + jj + (jj&4) + 4*(lane>=16)
// B from pre-packed global fragments: one contiguous 32B load per lane.
// ---------------------------------------------------------------------------
template <int KSTEPS>
__device__ inline v8f wave_gemm_p(const unsigned int* ldsA, const _Float16* wf,
                                  int mbase, int ntile, int lane) {
  v8f acc = {};
  const int m     = lane & 15;
  const int koff2 = (lane & 16) >> 2;   // +4 word-rows for upper half-lanes
#pragma unroll
  for (int ks = 0; ks < KSTEPS; ++ks) {
    union { v16h h; unsigned int u[8]; } af;
#pragma unroll
    for (int jj = 0; jj < 8; ++jj) {
      int wordRow = ks * 16 + jj + (jj & 4) + koff2;
      af.u[jj] = ldsA[wordRow * 64 + mbase + m];
    }
    v16h bf = *(const v16h*)(wf + (size_t)(((ntile * KSTEPS) + ks) * 32 + lane) * 16);
    acc = __builtin_amdgcn_wmma_f32_16x16x32_f16(false, af.h, false, bf,
                                                 (short)0, acc, false, false);
  }
  return acc;
}

// ---------------------------------------------------------------------------
// K1 / K5: out = conv1x1(w2,b2)( [silu] ln( conv1x1(w1,b1)(in) ) ), C=64
// grid: B*(HW/64) blocks of 256 threads (8 waves)
// ---------------------------------------------------------------------------
__global__ void conv_ln_conv_kernel(const float* __restrict__ in, float* __restrict__ out,
                                    const _Float16* __restrict__ w1f, const float* __restrict__ b1,
                                    const float* __restrict__ g,  const float* __restrict__ beta,
                                    const _Float16* __restrict__ w2f, const float* __restrict__ b2,
                                    int silu, int HW) {
  __shared__ unsigned int ldsA[32 * 64]; // packed f16 pairs [K/2][p]
  __shared__ float        ldsY[64 * 64]; // [p][o]
  const int tid = threadIdx.x, lane = tid & 31, wave = tid >> 5;
  const int tilesPerBatch = HW / PIX_TILE;
  const int b  = blockIdx.x / tilesPerBatch;
  const int p0 = (blockIdx.x % tilesPerBatch) * PIX_TILE;

  // stage input as packed f16 pairs
  for (int i = tid; i < 32 * 64; i += 256) {
    int q = i >> 6, p = i & 63;
    float v0 = in[(b * 64 + 2 * q    ) * HW + p0 + p];
    float v1 = in[(b * 64 + 2 * q + 1) * HW + p0 + p];
    ldsA[q * 64 + p] = pack_f16x2(v0, v1);
  }
  __syncthreads();

  // GEMM 1: y = in * w1^T + b1  -> ldsY
  for (int job = wave; job < 16; job += 8) {
    int mbase = (job & 3) * 16, ntile = (job >> 2);
    v8f acc = wave_gemm_p<2>(ldsA, w1f, mbase, ntile, lane);
    int n = lane & 15, mtop = (lane & 16) ? 8 : 0;
    float bias = b1[ntile * 16 + n];
#pragma unroll
    for (int r = 0; r < 8; ++r)
      ldsY[(mbase + r + mtop) * 64 + ntile * 16 + n] = acc[r] + bias;
  }
  __syncthreads();

  // LayerNorm over channels (+ optional SiLU), pack back into ldsA
  if (tid < 64) {
    int p = tid;
    float mean = 0.f;
    for (int c = 0; c < 64; ++c) mean += ldsY[p * 64 + c];
    mean *= (1.0f / 64.0f);
    float var = 0.f;
    for (int c = 0; c < 64; ++c) { float d = ldsY[p * 64 + c] - mean; var += d * d; }
    var *= (1.0f / 64.0f);
    float inv = rsqrtf(var + 1e-6f);
    for (int c = 0; c < 64; c += 2) {
      float v0 = g[c]     * ((ldsY[p * 64 + c]     - mean) * inv) + beta[c];
      float v1 = g[c + 1] * ((ldsY[p * 64 + c + 1] - mean) * inv) + beta[c + 1];
      if (silu) { v0 = v0 * sigmoidf_(v0); v1 = v1 * sigmoidf_(v1); }
      ldsA[(c >> 1) * 64 + p] = pack_f16x2(v0, v1);
    }
  }
  __syncthreads();

  // GEMM 2: out = y * w2^T + b2  -> global (vectorized: 8 consecutive pixels)
  for (int job = wave; job < 16; job += 8) {
    int mbase = (job & 3) * 16, ntile = (job >> 2);
    v8f acc = wave_gemm_p<2>(ldsA, w2f, mbase, ntile, lane);
    int n = lane & 15, mtop = (lane & 16) ? 8 : 0;
    int o = ntile * 16 + n;
    float bias = b2[o];
    float* dst = out + (size_t)(b * 64 + o) * HW + p0 + mbase + mtop;
    float4 lo = make_float4(acc[0] + bias, acc[1] + bias, acc[2] + bias, acc[3] + bias);
    float4 hi = make_float4(acc[4] + bias, acc[5] + bias, acc[6] + bias, acc[7] + bias);
    *(float4*)dst       = lo;
    *(float4*)(dst + 4) = hi;
  }
}

// ---------------------------------------------------------------------------
// K2: combined[b,n,h,w] = exp(-||px_nbr - px_ctr||^2 / 2C) * spatialGauss * valid
// grid: B*(H/16)*(W/16) blocks of 256 threads; halo tile in LDS, 2 ch-chunks
// ---------------------------------------------------------------------------
__global__ void bilateral_kernel(const float* __restrict__ px, const float* __restrict__ sigma_p,
                                 float* __restrict__ comb, int H, int W) {
  __shared__ float ldsH[32 * 22 * 22];
  const int tid = threadIdx.x;
  const int tilesX = W / 16;
  const int tilesPerBatch = (H / 16) * tilesX;
  const int b  = blockIdx.x / tilesPerBatch;
  const int t  = blockIdx.x % tilesPerBatch;
  const int ty = t / tilesX, tx = t % tilesX;
  const int py = tid >> 4, pxl = tid & 15;
  const int h = ty * 16 + py, w = tx * 16 + pxl;
  const int HW = H * W;

  float dist2[49];
#pragma unroll
  for (int n = 0; n < 49; ++n) dist2[n] = 0.f;

  for (int c0 = 0; c0 < 64; c0 += 32) {
    __syncthreads();
    for (int i = tid; i < 32 * 484; i += 256) {
      int c = i / 484, rr = i % 484;
      int hy = rr / 22, hx = rr % 22;
      int gh = ty * 16 - 3 + hy, gw = tx * 16 - 3 + hx;
      float v = 0.f;
      if (gh >= 0 && gh < H && gw >= 0 && gw < W)
        v = px[(b * 64 + c0 + c) * HW + gh * W + gw];
      ldsH[i] = v;
    }
    __syncthreads();
    for (int c = 0; c < 32; ++c) {
      const float* row = &ldsH[c * 484];
      float ctr = row[(py + 3) * 22 + (pxl + 3)];
      int n = 0;
#pragma unroll
      for (int dy = -3; dy <= 3; ++dy)
#pragma unroll
        for (int dx = -3; dx <= 3; ++dx, ++n) {
          float d = row[(py + 3 + dy) * 22 + (pxl + 3 + dx)] - ctr;
          dist2[n] += d * d;
        }
    }
  }

  float sig = sigma_p[0];
  float inv2s = 1.0f / (2.0f * sig * sig);
  int n = 0;
#pragma unroll
  for (int dy = -3; dy <= 3; ++dy)
#pragma unroll
    for (int dx = -3; dx <= 3; ++dx, ++n) {
      bool valid = (h + dy >= 0) && (h + dy < H) && (w + dx >= 0) && (w + dx < W);
      float sw = __expf(-(float)(dy * dy + dx * dx) * inv2s);
      float rw = __expf(-dist2[n] * (1.0f / 128.0f)); // 2*C = 128
      comb[(b * 49 + n) * HW + h * W + w] = valid ? rw * sw : 0.f;
    }
}

// ---------------------------------------------------------------------------
// K3: fixup — f = conv(49, [comb;sem](113)) -> LN(49) -> SiLU -> conv(49,49)
//     comb = normalize_n( comb * (1 + sigmoid(f)) )     (in-place on comb)
// K padded 113->128, N padded 49->64 (zeros baked into packed weights)
// ---------------------------------------------------------------------------
__global__ void fixup_kernel(float* __restrict__ comb, const float* __restrict__ sem,
                             const _Float16* __restrict__ w1f, const float* __restrict__ b1,
                             const float* __restrict__ g,  const float* __restrict__ beta,
                             const _Float16* __restrict__ w2f, const float* __restrict__ b2,
                             int HW) {
  __shared__ unsigned int ldsA[64 * 64]; // packed f16 pairs, 128 k-rows
  __shared__ float        ldsY[64 * 64]; // [p][n]
  const int tid = threadIdx.x, lane = tid & 31, wave = tid >> 5;
  const int tilesPerBatch = HW / PIX_TILE;
  const int b  = blockIdx.x / tilesPerBatch;
  const int p0 = (blockIdx.x % tilesPerBatch) * PIX_TILE;

  // stage [comb(49); sem(64); zeros(15)] as packed f16 pairs
  for (int i = tid; i < 64 * 64; i += 256) {
    int q = i >> 6, p = i & 63;
    int k0 = 2 * q, k1 = 2 * q + 1;
    float v0 = (k0 < 49) ? comb[(b * 49 + k0) * HW + p0 + p]
             : (k0 < 113 ? sem[(b * 64 + (k0 - 49)) * HW + p0 + p] : 0.f);
    float v1 = (k1 < 49) ? comb[(b * 49 + k1) * HW + p0 + p]
             : (k1 < 113 ? sem[(b * 64 + (k1 - 49)) * HW + p0 + p] : 0.f);
    ldsA[q * 64 + p] = pack_f16x2(v0, v1);
  }
  __syncthreads();

  // GEMM 1: K=128 (4 k-steps), N padded to 64
  for (int job = wave; job < 16; job += 8) {
    int mbase = (job & 3) * 16, ntile = (job >> 2);
    v8f acc = wave_gemm_p<4>(ldsA, w1f, mbase, ntile, lane);
    int n = lane & 15, mtop = (lane & 16) ? 8 : 0;
    int o = ntile * 16 + n;
    float bias = (o < 49) ? b1[o] : 0.f;
#pragma unroll
    for (int r = 0; r < 8; ++r)
      ldsY[(mbase + r + mtop) * 64 + o] = acc[r] + bias;
  }
  __syncthreads();

  // LN over 49 channels + SiLU, pack into ldsA rows 0..24 (rest zero)
  if (tid < 64) {
    int p = tid;
    float mean = 0.f;
    for (int c = 0; c < 49; ++c) mean += ldsY[p * 64 + c];
    mean *= (1.0f / 49.0f);
    float var = 0.f;
    for (int c = 0; c < 49; ++c) { float d = ldsY[p * 64 + c] - mean; var += d * d; }
    var *= (1.0f / 49.0f);
    float inv = rsqrtf(var + 1e-6f);
    for (int c = 0; c < 64; c += 2) {
      float v0 = 0.f, v1 = 0.f;
      if (c < 49) {
        float t0 = g[c] * ((ldsY[p * 64 + c] - mean) * inv) + beta[c];
        v0 = t0 * sigmoidf_(t0);
      }
      if (c + 1 < 49) {
        float t1 = g[c + 1] * ((ldsY[p * 64 + c + 1] - mean) * inv) + beta[c + 1];
        v1 = t1 * sigmoidf_(t1);
      }
      ldsA[(c >> 1) * 64 + p] = pack_f16x2(v0, v1);
    }
  }
  __syncthreads();

  // GEMM 2: K=64 (2 k-steps), 49x49 weights zero-padded in wf
  for (int job = wave; job < 16; job += 8) {
    int mbase = (job & 3) * 16, ntile = (job >> 2);
    v8f acc = wave_gemm_p<2>(ldsA, w2f, mbase, ntile, lane);
    int n = lane & 15, mtop = (lane & 16) ? 8 : 0;
    int o = ntile * 16 + n;
    float bias = (o < 49) ? b2[o] : 0.f;
#pragma unroll
    for (int r = 0; r < 8; ++r)
      ldsY[(mbase + r + mtop) * 64 + o] = acc[r] + bias;
  }
  __syncthreads();

  // gate + normalize over n2, in place on comb
  if (tid < 64) {
    int p = tid;
    float gvals[49];
    float s = 0.f;
#pragma unroll
    for (int n = 0; n < 49; ++n) {
      float c0 = comb[(b * 49 + n) * HW + p0 + p];
      float f  = ldsY[p * 64 + n];
      float gg = c0 * (1.0f + sigmoidf_(f));
      gvals[n] = gg;
      s += gg;
    }
    float invs = 1.0f / (s + 1e-7f);
#pragma unroll
    for (int n = 0; n < 49; ++n)
      comb[(b * 49 + n) * HW + p0 + p] = gvals[n] * invs;
  }
}

// ---------------------------------------------------------------------------
// K4: out0[b,c,h,w] = sum_n spatial[b,c,h+dy,w+dx] * comb[b,n,h,w]
// ---------------------------------------------------------------------------
__global__ void reduce_kernel(const float* __restrict__ spat, const float* __restrict__ comb,
                              float* __restrict__ out0, int H, int W) {
  __shared__ float ldsH[32 * 22 * 22];
  const int tid = threadIdx.x;
  const int tilesX = W / 16;
  const int tilesPerBatch = (H / 16) * tilesX;
  const int b  = blockIdx.x / tilesPerBatch;
  const int t  = blockIdx.x % tilesPerBatch;
  const int ty = t / tilesX, tx = t % tilesX;
  const int py = tid >> 4, pxl = tid & 15;
  const int h = ty * 16 + py, w = tx * 16 + pxl;
  const int HW = H * W;

  float cw[49];
#pragma unroll
  for (int n = 0; n < 49; ++n)
    cw[n] = comb[(b * 49 + n) * HW + h * W + w];

  for (int c0 = 0; c0 < 64; c0 += 32) {
    __syncthreads();
    for (int i = tid; i < 32 * 484; i += 256) {
      int c = i / 484, rr = i % 484;
      int hy = rr / 22, hx = rr % 22;
      int gh = ty * 16 - 3 + hy, gw = tx * 16 - 3 + hx;
      float v = 0.f;
      if (gh >= 0 && gh < H && gw >= 0 && gw < W)
        v = spat[(b * 64 + c0 + c) * HW + gh * W + gw];
      ldsH[i] = v;
    }
    __syncthreads();
    for (int c = 0; c < 32; ++c) {
      const float* row = &ldsH[c * 484];
      float acc = 0.f;
      int n = 0;
#pragma unroll
      for (int dy = -3; dy <= 3; ++dy)
#pragma unroll
        for (int dx = -3; dx <= 3; ++dx, ++n)
          acc += row[(py + 3 + dy) * 22 + (pxl + 3 + dx)] * cw[n];
      out0[(b * 64 + c0 + c) * HW + h * W + w] = acc;
    }
  }
}

// ---------------------------------------------------------------------------
extern "C" void kernel_launch(void* const* d_in, const int* in_sizes, int n_in,
                              void* d_out, int out_size, void* d_ws, size_t ws_size,
                              hipStream_t stream) {
  const float* spatial  = (const float*)d_in[0];
  const float* semantic = (const float*)d_in[1];
  const float* rp_w1 = (const float*)d_in[2];
  const float* rp_b1 = (const float*)d_in[3];
  const float* rp_g  = (const float*)d_in[4];
  const float* rp_be = (const float*)d_in[5];
  const float* rp_w2 = (const float*)d_in[6];
  const float* rp_b2 = (const float*)d_in[7];
  const float* fx_w1 = (const float*)d_in[8];
  const float* fx_b1 = (const float*)d_in[9];
  const float* fx_g  = (const float*)d_in[10];
  const float* fx_be = (const float*)d_in[11];
  const float* fx_w2 = (const float*)d_in[12];
  const float* fx_b2 = (const float*)d_in[13];
  const float* op_w1 = (const float*)d_in[14];
  const float* op_b1 = (const float*)d_in[15];
  const float* op_g  = (const float*)d_in[16];
  const float* op_be = (const float*)d_in[17];
  const float* op_w2 = (const float*)d_in[18];
  const float* op_b2 = (const float*)d_in[19];
  const float* sigma = (const float*)d_in[20];

  const int B = 2, C = 64, H = 112, W = 112, HW = H * W, N2 = 49;

  float* ws   = (float*)d_ws;
  float* px   = ws;                         // B*C*HW
  float* comb = px + (size_t)B * C * HW;    // B*N2*HW
  float* out0 = comb + (size_t)B * N2 * HW; // B*C*HW

  // packed f16 weight fragments (all offsets 32B aligned)
  _Float16* wpack  = (_Float16*)(out0 + (size_t)B * C * HW);
  _Float16* rp_w1f = wpack;            // 4 ntiles * 2 ksteps * 512 = 4096
  _Float16* rp_w2f = rp_w1f + 4096;
  _Float16* op_w1f = rp_w2f + 4096;
  _Float16* op_w2f = op_w1f + 4096;
  _Float16* fx_w1f = op_w2f + 4096;    // 4 ntiles * 4 ksteps * 512 = 8192
  _Float16* fx_w2f = fx_w1f + 8192;    // 4096

  // 0. pre-pack weights into WMMA B-fragment order (f16)
  pack_weights_kernel<<<16, 256, 0, stream>>>(rp_w1, rp_w1f, 64, 64, 2, 4);
  pack_weights_kernel<<<16, 256, 0, stream>>>(rp_w2, rp_w2f, 64, 64, 2, 4);
  pack_weights_kernel<<<16, 256, 0, stream>>>(op_w1, op_w1f, 64, 64, 2, 4);
  pack_weights_kernel<<<16, 256, 0, stream>>>(op_w2, op_w2f, 64, 64, 2, 4);
  pack_weights_kernel<<<16, 256, 0, stream>>>(fx_w1, fx_w1f, 113, 49, 4, 4);
  pack_weights_kernel<<<16, 256, 0, stream>>>(fx_w2, fx_w2f, 49, 49, 2, 4);

  const int gemmBlocks = B * (HW / PIX_TILE);     // 392
  const int tileBlocks = B * (H / 16) * (W / 16); // 98

  // 1. range_proj(semantic) -> px  (WMMA GEMMs + LN + SiLU)
  conv_ln_conv_kernel<<<gemmBlocks, 256, 0, stream>>>(
      semantic, px, rp_w1f, rp_b1, rp_g, rp_be, rp_w2f, rp_b2, /*silu=*/1, HW);

  // 2. bilateral combined kernel
  bilateral_kernel<<<tileBlocks, 256, 0, stream>>>(px, sigma, comb, H, W);

  // 3. gated fixup (WMMA GEMMs, in-place on comb)
  fixup_kernel<<<gemmBlocks, 256, 0, stream>>>(
      comb, semantic, fx_w1f, fx_b1, fx_g, fx_be, fx_w2f, fx_b2, HW);

  // 4. neighborhood-weighted reduction of spatial feats
  reduce_kernel<<<tileBlocks, 256, 0, stream>>>(spatial, comb, out0, H, W);

  // 5. output_proj -> d_out (WMMA GEMMs + LN, no SiLU)
  conv_ln_conv_kernel<<<gemmBlocks, 256, 0, stream>>>(
      out0, (float*)d_out, op_w1f, op_b1, op_g, op_be, op_w2f, op_b2, /*silu=*/0, HW);
}
